// TensorFusion_82463372083295
// MI455X (gfx1250) — matrix operand order repaired and checked by reading, not compiled, not run on hardware
//
#include <hip/hip_runtime.h>
#include <hip/hip_bf16.h>

// ---------------- problem constants ----------------
#define BT      64          // B*T = 2*32
#define SPK     64          // spike/token axis S
#define DV      768
#define DA      512
#define SH      512         // SHARED
#define AD      513         // SHARED+1
#define KTOT    (AD*AD)     // 263169
#define EPAD    544         // 17 * 32  (padded e dimension)
#define NOUT1   1024        // 2*SHARED
#define DECAY   0.9f
#define THRESH  1.0f

typedef __attribute__((ext_vector_type(16))) __bf16 v16bf;
typedef __attribute__((ext_vector_type(8)))  float  v8f;

// ---------------- tiny front-end kernels ----------------

// mean over the spike axis: x (BT, S, D) -> p (BT, D)
__global__ void pool_kernel(const float* __restrict__ x, float* __restrict__ p,
                            int D) {
  int idx = blockIdx.x * blockDim.x + threadIdx.x;
  if (idx >= BT * D) return;
  int bt = idx / D, d = idx - bt * D;
  const float* base = x + ((size_t)bt * SPK) * D + d;
  float s = 0.f;
  #pragma unroll 4
  for (int i = 0; i < SPK; ++i) s += base[(size_t)i * D];
  p[idx] = s * (1.0f / SPK);
}

// y[r, o] = (relu?) (x[r,:] . W[o,:] + bias[o]); y row stride = ostride
__global__ void linear_kernel(const float* __restrict__ x, const float* __restrict__ W,
                              const float* __restrict__ bias, float* __restrict__ y,
                              int R, int K, int O, int relu, int ostride) {
  int idx = blockIdx.x * blockDim.x + threadIdx.x;
  if (idx >= R * O) return;
  int r = idx / O, o = idx - r * O;
  const float* xr = x + (size_t)r * K;
  const float* wr = W + (size_t)o * K;
  float acc = bias[o];
  #pragma unroll 4
  for (int k = 0; k < K; ++k) acc = fmaf(xr[k], wr[k], acc);
  if (relu) acc = fmaxf(acc, 0.f);
  y[(size_t)r * ostride + o] = acc;
}

// append the ones column of the (BT, AD) a/b matrices
__global__ void ones_kernel(float* __restrict__ a, float* __restrict__ b) {
  int r = threadIdx.x;
  if (r < BT) {
    a[(size_t)r * AD + (AD - 1)] = 1.0f;
    b[(size_t)r * AD + (AD - 1)] = 1.0f;
  }
}

__global__ void zero_kernel(float* __restrict__ p, int n) {
  int i = blockIdx.x * blockDim.x + threadIdx.x;
  if (i < n) p[i] = 0.f;
}

// ---------------- the big bilinear-fusion GEMM (WMMA bf16) ----------------
// H(64,1024) += tp(64,263169) @ fw1^T, tp[m, d*513+e] = a[m,d]*b[m,e]
// grid: (x = 1024/64 = 16 N-groups, y = 4 M-tiles, z = 8 K-splits over d)
// block: 128 threads = 4 waves; wave w handles N-tile (4*blockIdx.x + w).
__global__ __launch_bounds__(128) void fusion1_wmma(
    const float* __restrict__ Amat,   // (BT, AD)
    const float* __restrict__ Bmat,   // (BT, AD)
    const float* __restrict__ W1,     // (NOUT1, KTOT) row-major
    float* __restrict__ H)            // (BT, NOUT1) accumulated atomically
{
  __shared__ float bl[16 * EPAD];     // b-tile, fp32, e padded with zeros

  const int tid   = threadIdx.x;
  const int wave  = tid >> 5;
  const int lane  = tid & 31;
  const int mtile = blockIdx.y;                 // 0..3
  const int ntile = blockIdx.x * 4 + wave;      // 0..63
  const int nz    = gridDim.z;
  const int dper  = (AD + nz - 1) / nz;
  const int d0    = blockIdx.z * dper;
  int d1 = d0 + dper; if (d1 > AD) d1 = AD;

  // cooperative stage of this M-tile's b rows into LDS (zero-padded to EPAD)
  for (int i = tid; i < 16 * EPAD; i += 128) {
    int mm = i / EPAD, e = i - mm * EPAD;
    bl[i] = (e < AD) ? Bmat[(size_t)(mtile * 16 + mm) * AD + e] : 0.0f;
  }
  __syncthreads();

  const int m     = lane & 15;        // A-row / B-column within tile
  const int half  = lane >> 4;        // lane half selects K sub-ranges
  const int gm    = mtile * 16 + m;
  const int kA0   = half * 8;         // A frag: j<8 -> K=kA0+j ; j>=8 -> 16+kA0+(j-8)
  const int kB0   = half * 16;        // prefetch offset only; B frag uses A's K order (kA0)
  const float* __restrict__ wrow = W1 + (size_t)(ntile * 16 + m) * (size_t)KTOT;
  const float* __restrict__ blm  = bl + m * EPAD;

  v8f c = {0.f, 0.f, 0.f, 0.f, 0.f, 0.f, 0.f, 0.f};

  for (int d = d0; d < d1; ++d) {
    const float asc = Amat[(size_t)gm * AD + d];      // per-lane scalar a[m,d]
    const float* __restrict__ wd = wrow + (size_t)d * AD;

    // -------- main 16 K-blocks: e in [0, 512), branch-free --------
    #pragma unroll 2
    for (int blk = 0; blk < 16; ++blk) {
      const int e0 = blk * 32;

      // A fragment: bf16( a[m,d] * b[m, e0 + K] )
      v16bf af;
      #pragma unroll
      for (int j = 0; j < 8; ++j) {
        af[j]     = (__bf16)(asc * blm[e0 + kA0 + j]);
        af[j + 8] = (__bf16)(asc * blm[e0 + 16 + kA0 + j]);
      }

      // B fragment: bf16( fw1[o, d*513 + e0 + K] ), 4x b128 stream
      const int eB = e0 + kB0;
      __builtin_prefetch(wd + eB + 96, 0, 1);         // stream weights ahead
      v16bf bf;
      #pragma unroll
      for (int j = 0; j < 16; ++j) bf[j] = (__bf16)wd[e0 + (j < 8 ? kA0 + j : 16 + kA0 + (j - 8))];

      c = __builtin_amdgcn_wmma_f32_16x16x32_bf16(
              false, af, false, bf, (short)0, c, false, false);
    }

    // -------- tail K-block (e = 512..543): only e==512 (ones col) is live ---
    {
      // A fragment is zero except K=0 (low lane-half, j==0): value = a[m,d]*1
      v16bf af = (v16bf)(__bf16)0.0f;
      af[0] = (__bf16)(half ? 0.0f : asc);
      // B fragment: only K=0 matters -> broadcast fw1[o, d*513 + 512]
      const __bf16 wb = (__bf16)wd[SH];
      v16bf bf;
      #pragma unroll
      for (int j = 0; j < 16; ++j) bf[j] = wb;

      c = __builtin_amdgcn_wmma_f32_16x16x32_bf16(
              false, af, false, bf, (short)0, c, false, false);
    }
  }

  // C/D layout: element r -> (M = half*8 + r, N = m); commit K-split partials
  #pragma unroll
  for (int r = 0; r < 8; ++r) {
    int row = mtile * 16 + half * 8 + r;
    atomicAdd(&H[(size_t)row * NOUT1 + ntile * 16 + m], c[r]);
  }
}

// ---------------- epilogue: relu(H+fb1) @ fw2^T + fb2 ----------------
__global__ void fuse2_kernel(const float* __restrict__ H, const float* __restrict__ fb1,
                             const float* __restrict__ W2, const float* __restrict__ fb2,
                             float* __restrict__ fused) {
  int idx = blockIdx.x * blockDim.x + threadIdx.x;
  if (idx >= BT * SH) return;
  int r = idx / SH, o = idx - r * SH;
  const float* hr = H + (size_t)r * NOUT1;
  const float* wr = W2 + (size_t)o * NOUT1;
  float acc = fb2[o];
  #pragma unroll 4
  for (int k = 0; k < NOUT1; ++k)
    acc = fmaf(fmaxf(hr[k] + fb1[k], 0.f), wr[k], acc);
  fused[(size_t)r * SH + o] = acc;
}

// ---------------- LIF over the time axis ----------------
__global__ void lif_kernel(const float* __restrict__ fused, float* __restrict__ out) {
  int idx = blockIdx.x * blockDim.x + threadIdx.x;   // (b, f), 2*512 = 1024
  if (idx >= 2 * SH) return;
  int b = idx / SH, f = idx - b * SH;
  float mem = 0.f;
  for (int t = 0; t < 32; ++t) {
    size_t off = ((size_t)b * 32 + t) * SH + f;
    mem = DECAY * mem + fused[off];
    float s = (mem - THRESH >= 0.f) ? 1.f : 0.f;
    out[off] = s;
    mem -= s * THRESH;
  }
}

// ---------------- launch ----------------
extern "C" void kernel_launch(void* const* d_in, const int* in_sizes, int n_in,
                              void* d_out, int out_size, void* d_ws, size_t ws_size,
                              hipStream_t stream) {
  const float* vision = (const float*)d_in[0];
  const float* audio  = (const float*)d_in[1];
  const float* vw1 = (const float*)d_in[2];  const float* vb1 = (const float*)d_in[3];
  const float* vw2 = (const float*)d_in[4];  const float* vb2 = (const float*)d_in[5];
  const float* aw1 = (const float*)d_in[6];  const float* ab1 = (const float*)d_in[7];
  const float* aw2 = (const float*)d_in[8];  const float* ab2 = (const float*)d_in[9];
  const float* fw1 = (const float*)d_in[10]; const float* fb1 = (const float*)d_in[11];
  const float* fw2 = (const float*)d_in[12]; const float* fb2 = (const float*)d_in[13];
  float* out = (float*)d_out;

  // workspace layout (floats), ~1.25 MB total
  float* ws    = (float*)d_ws;
  float* p_v   = ws;                    // 64*768
  float* p_a   = p_v  + BT * DV;        // 64*512
  float* h_v   = p_a  + BT * DA;        // 64*512
  float* h_a   = h_v  + BT * SH;        // 64*512
  float* amat  = h_a  + BT * SH;        // 64*513
  float* bmat  = amat + BT * AD;        // 64*513
  float* Hbuf  = bmat + BT * AD;        // 64*1024
  float* fused = Hbuf + BT * NOUT1;     // 64*512

  const int TB = 256;
  pool_kernel<<<(BT*DV + TB-1)/TB, TB, 0, stream>>>(vision, p_v, DV);
  pool_kernel<<<(BT*DA + TB-1)/TB, TB, 0, stream>>>(audio,  p_a, DA);

  linear_kernel<<<(BT*SH + TB-1)/TB, TB, 0, stream>>>(p_v, vw1, vb1, h_v, BT, DV, SH, 1, SH);
  linear_kernel<<<(BT*SH + TB-1)/TB, TB, 0, stream>>>(h_v, vw2, vb2, amat, BT, SH, SH, 0, AD);
  linear_kernel<<<(BT*SH + TB-1)/TB, TB, 0, stream>>>(p_a, aw1, ab1, h_a, BT, DA, SH, 1, SH);
  linear_kernel<<<(BT*SH + TB-1)/TB, TB, 0, stream>>>(h_a, aw2, ab2, bmat, BT, SH, SH, 0, AD);
  ones_kernel<<<1, BT, 0, stream>>>(amat, bmat);

  zero_kernel<<<(BT*NOUT1 + TB-1)/TB, TB, 0, stream>>>(Hbuf, BT*NOUT1);

  // 16 N-groups x 4 M-tiles x 8 K-splits, 4 waves/block
  fusion1_wmma<<<dim3(16, 4, 8), 128, 0, stream>>>(amat, bmat, fw1, Hbuf);

  fuse2_kernel<<<(BT*SH + TB-1)/TB, TB, 0, stream>>>(Hbuf, fb1, fw2, fb2, fused);
  lif_kernel<<<(2*SH + TB-1)/TB, TB, 0, stream>>>(fused, out);
}
